// Soft_MoE_32925219291187
// MI455X (gfx1250) — matrix operand order repaired and checked by reading, not compiled, not run on hardware
//
#include <hip/hip_runtime.h>
#include <math.h>
#include <stdint.h>

// ---------------- problem constants ----------------
constexpr int BN_  = 8;     // batch
constexpr int C_   = 256;   // channels
constexpr int H_   = 128, W_ = 128;
constexpr int NH_ = 4, P_ = 4;
constexpr int HID_ = 64;
constexpr int L_   = H_ * W_;        // 16384
constexpr int LP_  = (H_/P_)*(W_/P_);// 1024
constexpr int CG_  = C_ / 4;         // 64
constexpr int DH_  = LP_ / NH_;      // 256  (pooled cols per head)
constexpr int DV_  = L_ / NH_;       // 4096 (full-res cols per head)

typedef __attribute__((ext_vector_type(16))) __bf16 v16bf;
typedef __attribute__((ext_vector_type(8)))  __bf16 v8bf;
typedef __attribute__((ext_vector_type(8)))  float  v8f;
typedef __attribute__((ext_vector_type(4)))  unsigned int u32x4;
typedef __attribute__((ext_vector_type(8)))  int i32x8;
typedef __attribute__((ext_vector_type(4)))  int i32x4;

// ---------------------------------------------------------------------------
// LDS layouts: every WMMA fragment is ONE contiguous 32B read (2x ds_load_b128)
// in contiguous k order (k = 16*(lane>>4) + s) for both A and B.
//  A: per row of 32 k, 8-elem chunks stored [K0-7|K8-15|K16-23|K24-31],
//     row stride 48 elems; lane group (lane>>4) reads 16 elems at offset 0/16.
//  B: stored [n][k] (k contiguous), row stride 48; lane reads K=kb..kb+15.
// Double-buffered (ping-pong): buffer stride 3072 elems; ONE barrier per
// K-step — stage tile k+1 into alt buffer while WMMA consumes tile k.
// ---------------------------------------------------------------------------

template <typename TB, bool BT>
__device__ __forceinline__ void stageAB(const __bf16* __restrict__ Ag, int lda,
                                        const TB* __restrict__ Bg, int ldb,
                                        int k0, __bf16* As, __bf16* Bn, int tid) {
  // ---- A: 64 rows x 4 chunks of 8 bf16, chunk->slot identity (vec/vec) ----
#pragma unroll
  for (int it = 0; it < 2; ++it) {
    int idx = tid + it * 128;
    int r = idx >> 2, c = idx & 3;
    int slot = c;  // natural k order: slot == chunk
    *(uint4*)(As + r * 48 + slot * 8) =
        *(const uint4*)(Ag + (size_t)r * lda + k0 + c * 8);
  }
  // ---- B into [n][k] ----
  if constexpr (BT) {  // source already [n][k]: vector/vector
#pragma unroll
    for (int it = 0; it < 2; ++it) {
      int idx = tid + it * 128;
      int n = idx >> 2, c = (idx & 3) * 8;
      *(uint4*)(Bn + n * 48 + c) =
          *(const uint4*)((const __bf16*)Bg + (size_t)n * ldb + k0 + c);
    }
  } else if constexpr (sizeof(TB) == 2) {  // [k][n] bf16: k-pair packed b32
    int kp = tid >> 3, nc = (tid & 7) * 8;  // 128 chunks == 128 threads
    const __bf16* s0 = (const __bf16*)Bg + (size_t)(k0 + 2 * kp) * ldb + nc;
    v8bf d0 = *(const v8bf*)s0;
    v8bf d1 = *(const v8bf*)(s0 + ldb);
#pragma unroll
    for (int e = 0; e < 8; ++e) {
      union { __bf16 h[2]; unsigned u; } p;
      p.h[0] = d0[e]; p.h[1] = d1[e];
      *(unsigned*)(Bn + (nc + e) * 48 + 2 * kp) = p.u;
    }
  } else {  // [k][n] f32: convert + k-pair packed b32
    int kp = tid >> 3, nc = (tid & 7) * 8;
    const float* s0 = (const float*)Bg + (size_t)(k0 + 2 * kp) * ldb + nc;
#pragma unroll
    for (int e = 0; e < 8; ++e) {
      union { __bf16 h[2]; unsigned u; } p;
      p.h[0] = (__bf16)s0[e]; p.h[1] = (__bf16)s0[ldb + e];
      *(unsigned*)(Bn + (nc + e) * 48 + 2 * kp) = p.u;
    }
  }
}

// 64x64 block-tile GEMM core, K multiple of 32, 128 threads (4 waves).
// Wave w owns rows [16w,16w+16); acc[4] = 4 n-tiles of 16.
template <typename TB, bool BT>
__device__ void gemm64x64(const __bf16* __restrict__ Ag, int lda,
                          const TB* __restrict__ Bg, int ldb, int K,
                          v8f* acc, __bf16* As, __bf16* Bn) {
  const int tid  = threadIdx.x;
  const int lane = tid & 31;
  const int wave = tid >> 5;
  const int KT = K >> 5;
  stageAB<TB, BT>(Ag, lda, Bg, ldb, 0, As, Bn, tid);
  for (int kt = 0; kt < KT; ++kt) {
    __syncthreads();
    int cur = (kt & 1) * 3072, nxt = ((kt + 1) & 1) * 3072;
    if (kt + 1 < KT)
      stageAB<TB, BT>(Ag, lda, Bg, ldb, (kt + 1) << 5, As + nxt, Bn + nxt, tid);
    const __bf16* Ac = As + cur;
    const __bf16* Bc = Bn + cur;
    v16bf af = *(const v16bf*)(Ac + (wave * 16 + (lane & 15)) * 48 + ((lane >> 4) << 4));
#pragma unroll
    for (int nt = 0; nt < 4; ++nt) {
      v16bf bfv = *(const v16bf*)(Bc + (nt * 16 + (lane & 15)) * 48 + ((lane >> 4) << 4));
      acc[nt] = __builtin_amdgcn_wmma_f32_16x16x32_bf16(
          false, af, false, bfv, (short)0, acc[nt], false, false);
    }
  }
  __syncthreads();  // LDS safe for reuse by caller
}

__device__ inline int frag_row(int wave, int lane, int r) {
  return wave * 16 + ((lane >> 4) << 3) + r;
}
__device__ inline int frag_col(int nt, int lane) { return nt * 16 + (lane & 15); }

// ---------------- small kernels ----------------
__global__ void k_f32bf16(const float* __restrict__ s, __bf16* __restrict__ d, int n) {
  int i = blockIdx.x * blockDim.x + threadIdx.x;
  if (i < n) d[i] = (__bf16)s[i];
}

__global__ void k_bias(const float* __restrict__ w1, const float* __restrict__ b1,
                       const float* __restrict__ w2, float* __restrict__ bias) {
  int i = blockIdx.x, j = threadIdx.x;
  float rel = (float)(j - i) / (float)(C_ - 1) * 8.f;
  float a = fabsf(rel);
  float sgn = (rel > 0.f) ? 1.f : (rel < 0.f ? -1.f : 0.f);
  float f = sgn * log2f(a + 1.f) * (1.f / 3.f);  // /log2(8)
  float o0 = 0.f, o1 = 0.f, o2 = 0.f, o3 = 0.f;
#pragma unroll 8
  for (int t = 0; t < HID_; ++t) {
    float hv = fmaxf(f * w1[t] + b1[t], 0.f);
    o0 += hv * w2[t * NH_ + 0]; o1 += hv * w2[t * NH_ + 1];
    o2 += hv * w2[t * NH_ + 2]; o3 += hv * w2[t * NH_ + 3];
  }
  int ij = i * C_ + j;
  bias[0 * 65536 + ij] = 1.f / (1.f + expf(-o0));
  bias[1 * 65536 + ij] = 1.f / (1.f + expf(-o1));
  bias[2 * 65536 + ij] = 1.f / (1.f + expf(-o2));
  bias[3 * 65536 + ij] = 1.f / (1.f + expf(-o3));
}

__global__ void k_maxpool(const float* __restrict__ x, float* __restrict__ xp) {
  int idx = blockIdx.x * blockDim.x + threadIdx.x;  // B*C*LP
  int p = idx & (LP_ - 1);
  int c = (idx >> 10) & (C_ - 1);
  int b = idx >> 18;
  int py = p >> 5, px = p & 31;
  const float* src = x + ((size_t)(b * C_ + c) << 14) + (py * 4) * W_ + px * 4;
  float m = -3.4e38f;
#pragma unroll
  for (int dy = 0; dy < 4; ++dy) {
    float4 r0 = *(const float4*)(src + dy * W_);
    m = fmaxf(m, fmaxf(fmaxf(r0.x, r0.y), fmaxf(r0.z, r0.w)));
  }
  xp[idx] = m;
}

__global__ void k_qk(const float* __restrict__ xp,
                     const float* __restrict__ qw, const float* __restrict__ qb,
                     const float* __restrict__ kw, const float* __restrict__ kb,
                     float* __restrict__ q, float* __restrict__ k) {
  int idx = blockIdx.x * blockDim.x + threadIdx.x;  // B*C*LP
  int l = idx & (LP_ - 1);
  int c = (idx >> 10) & (C_ - 1);
  int b = idx >> 18;
  int g = c >> 6, i = c & 63;
  const float* xb = xp + (((size_t)b * C_ + g * CG_) << 10) + l;
  const float* wq = qw + (g * CG_ + i) * CG_;
  const float* wk = kw + (g * CG_ + i) * CG_;
  float aq = 0.f, ak = 0.f;
#pragma unroll 8
  for (int j = 0; j < CG_; ++j) {
    float xv = xb[(size_t)j << 10];
    aq = fmaf(wq[j], xv, aq);
    ak = fmaf(wk[j], xv, ak);
  }
  q[idx] = aq + qb[c];
  k[idx] = ak + kb[c];
}

__global__ void k_rownorm(const float* __restrict__ q, const float* __restrict__ k,
                          __bf16* __restrict__ qn, __bf16* __restrict__ kn) {
  int wid = (blockIdx.x * blockDim.x + threadIdx.x) >> 5;
  int lane = threadIdx.x & 31;
  if (wid >= BN_ * NH_ * C_) return;
  int c = wid & 255, h = (wid >> 8) & 3, b = wid >> 10;
  size_t src = (((size_t)b * C_ + c) << 10) + h * DH_;
  size_t dst = ((size_t)((b * NH_ + h) * C_ + c)) << 8;
  for (int pass = 0; pass < 2; ++pass) {
    const float* s = (pass == 0 ? q : k) + src;
    __bf16* d = (pass == 0 ? qn : kn) + dst;
    float v[8]; float ss = 0.f;
#pragma unroll
    for (int t = 0; t < 8; ++t) { v[t] = s[lane + 32 * t]; ss = fmaf(v[t], v[t], ss); }
#pragma unroll
    for (int m = 16; m; m >>= 1) ss += __shfl_xor(ss, m, 32);
    float inv = 1.f / fmaxf(sqrtf(ss), 1e-12f);
#pragma unroll
    for (int t = 0; t < 8; ++t) d[lane + 32 * t] = (__bf16)(v[t] * inv);
  }
}

// sim = (q̂ k̂ᵀ) * scale + bias  (32 batches of 256x256x256, bf16 WMMA, BT path)
__global__ void k_gemm_sim(const __bf16* __restrict__ qn, const __bf16* __restrict__ kn,
                           const float* __restrict__ bias, const float* __restrict__ ls,
                           float* __restrict__ sim) {
  __shared__ __attribute__((aligned(32))) __bf16 As[2 * 3072];
  __shared__ __attribute__((aligned(32))) __bf16 Bn[2 * 3072];
  int bid = blockIdx.x;
  int nb = bid & 3, mb = (bid >> 2) & 3, bh = bid >> 4;
  int h = bh & 3;
  v8f acc[4] = {};
  gemm64x64<__bf16, true>(qn + (size_t)bh * 65536 + mb * 64 * 256, 256,
                          kn + (size_t)bh * 65536 + nb * 64 * 256, 256, 256,
                          acc, As, Bn);
  int lane = threadIdx.x & 31, wave = threadIdx.x >> 5;
  float scale = expf(fminf(ls[h], 4.6051702f));  // log(100)
#pragma unroll
  for (int nt = 0; nt < 4; ++nt)
#pragma unroll
    for (int r = 0; r < 8; ++r) {
      int row = mb * 64 + frag_row(wave, lane, r);
      int col = nb * 64 + frag_col(nt, lane);
      sim[(size_t)bh * 65536 + row * 256 + col] =
          acc[nt][r] * scale + bias[(size_t)h * 65536 + row * 256 + col];
    }
}

__global__ void k_softmax2(const float* __restrict__ sim, __bf16* __restrict__ attn) {
  int wid = (blockIdx.x * blockDim.x + threadIdx.x) >> 5;
  int lane = threadIdx.x & 31;
  if (wid >= BN_ * NH_ * C_) return;
  int c = wid & 255, bh = wid >> 8;
  const float* s = sim + ((size_t)bh << 16) + ((size_t)c << 8);
  float v[8]; float mx = -3.4e38f;
#pragma unroll
  for (int t = 0; t < 8; ++t) { v[t] = s[lane + 32 * t]; mx = fmaxf(mx, v[t]); }
#pragma unroll
  for (int m = 16; m; m >>= 1) mx = fmaxf(mx, __shfl_xor(mx, m, 32));
  float sum = 0.f;
#pragma unroll
  for (int t = 0; t < 8; ++t) { v[t] = expf(v[t] - mx); sum += v[t]; }
#pragma unroll
  for (int m = 16; m; m >>= 1) sum += __shfl_xor(sum, m, 32);
  float inv = 1.f / sum;
#pragma unroll
  for (int t = 0; t < 8; ++t) v[t] = 1.f - v[t] * inv;  // 1 - softmax
  mx = -3.4e38f;
#pragma unroll
  for (int t = 0; t < 8; ++t) mx = fmaxf(mx, v[t]);
#pragma unroll
  for (int m = 16; m; m >>= 1) mx = fmaxf(mx, __shfl_xor(mx, m, 32));
  sum = 0.f;
#pragma unroll
  for (int t = 0; t < 8; ++t) { v[t] = expf(v[t] - mx); sum += v[t]; }
#pragma unroll
  for (int m = 16; m; m >>= 1) sum += __shfl_xor(sum, m, 32);
  inv = 1.f / sum;
  __bf16* d = attn + ((size_t)bh << 16) + ((size_t)c << 8);
#pragma unroll
  for (int t = 0; t < 8; ++t) d[lane + 32 * t] = (__bf16)(v[t] * inv);
}

__global__ void k_dwconv(const float* __restrict__ x, const float* __restrict__ w,
                         const float* __restrict__ b, __bf16* __restrict__ vd) {
  int idx = blockIdx.x * blockDim.x + threadIdx.x;  // B*C*L
  int hw = idx & (L_ - 1);
  int c = (idx >> 14) & (C_ - 1);
  int bb = idx >> 22;
  int y = hw >> 7, xx = hw & 127;
  const float* wc = w + c * 9;
  const float* src = x + ((size_t)(bb * C_ + c) << 14);
  float s = 0.f;
#pragma unroll
  for (int dy = -1; dy <= 1; ++dy) {
    int yy = y + dy;
    if (yy < 0 || yy >= H_) continue;
#pragma unroll
    for (int dx = -1; dx <= 1; ++dx) {
      int xc = xx + dx;
      if (xc < 0 || xc >= W_) continue;
      s = fmaf(wc[(dy + 1) * 3 + (dx + 1)], src[(yy << 7) + xc], s);
    }
  }
  vd[idx] = (__bf16)fmaxf(s + b[c], 0.f);
}

// grouped 1x1 on vd as 32 x (64x16384x64) bf16 WMMA GEMMs
__global__ void k_gemm_v(const __bf16* __restrict__ wpw, const __bf16* __restrict__ vd,
                         const float* __restrict__ pb, __bf16* __restrict__ v) {
  __shared__ __attribute__((aligned(32))) __bf16 As[2 * 3072];
  __shared__ __attribute__((aligned(32))) __bf16 Bn[2 * 3072];
  int bid = blockIdx.x;
  int nb = bid & 255, bg = bid >> 8;
  int b = bg >> 2, g = bg & 3;
  v8f acc[4] = {};
  gemm64x64<__bf16, false>(wpw + g * CG_ * CG_, CG_,
                           vd + (((size_t)b * C_ + g * CG_) << 14) + nb * 64, L_,
                           CG_, acc, As, Bn);
  int lane = threadIdx.x & 31, wave = threadIdx.x >> 5;
#pragma unroll
  for (int nt = 0; nt < 4; ++nt)
#pragma unroll
    for (int r = 0; r < 8; ++r) {
      int c = g * CG_ + frag_row(wave, lane, r);
      int col = nb * 64 + frag_col(nt, lane);
      v[(((size_t)b * C_ + c) << 14) + col] = (__bf16)(acc[nt][r] + pb[c]);
    }
}

// out = attn @ vh : 32 batches of 256x4096x256.  A tile staged by the Tensor
// Data Mover (double-buffered, one tensor_load_to_lds in flight); B transposed
// into LDS manually.
__global__ void k_gemm_attnout(const __bf16* __restrict__ attn, const __bf16* __restrict__ v,
                               __bf16* __restrict__ ao) {
  __shared__ __attribute__((aligned(64))) __bf16 Asp[2 * 2048];  // plain [m][k]
  __shared__ __attribute__((aligned(32))) __bf16 Bn[2 * 3072];   // [n][k]
  int bid = blockIdx.x;
  int nb = bid & 63, mb = (bid >> 6) & 3, bh = bid >> 8;
  int b = bh >> 2, h = bh & 3;
  const __bf16* Ap = attn + (size_t)bh * 65536 + mb * 64 * 256;
  const __bf16* Bp = v + (((size_t)b * C_) << 14) + h * DV_ + nb * 64;
  const int tid = threadIdx.x, lane = tid & 31, wave = tid >> 5;

  auto stage = [&](int k0, __bf16* Ad, __bf16* Bd) {
#if __has_builtin(__builtin_amdgcn_tensor_load_to_lds)
    if (wave == 0) {
      // D# group0: count=1 | lds_addr | global_addr | type=2 ("image")
      unsigned long long ga = (unsigned long long)(uintptr_t)(Ap + k0);
      unsigned lds = (unsigned)(uintptr_t)Ad;
      u32x4 g0 = {1u, lds, (unsigned)ga,
                  (unsigned)(((ga >> 32) & 0x01FFFFFFull) | 0x80000000ull)};
      // group1: data_size=2B; tensor 256x256 stride 256; tile 32(k) x 64(m)
      i32x8 g1 = {0x00010000,   // workgroup_mask=0, data_size=1 (2 bytes)
                  0x01000000,   // tensor_dim0 lo16 (256) << 16
                  0x01000000,   // tensor_dim0 hi | tensor_dim1 lo16 (256) << 16
                  0x00200000,   // tensor_dim1 hi | tile_dim0 (32) << 16
                  64,           // tile_dim1=64, tile_dim2=0
                  256,          // tensor_dim0_stride lo32
                  0, 0};
      i32x4 z4 = {0, 0, 0, 0};
#if __clang_major__ >= 23
      i32x8 z8 = {0, 0, 0, 0, 0, 0, 0, 0};
      __builtin_amdgcn_tensor_load_to_lds(g0, g1, z4, z4, z8, 0);
#else
      __builtin_amdgcn_tensor_load_to_lds(g0, g1, z4, z4, 0);
#endif
    }
#else
#pragma unroll
    for (int it = 0; it < 2; ++it) {
      int idx = tid + it * 128;
      int r = idx >> 2, c = (idx & 3) * 8;
      *(uint4*)(Ad + r * 32 + c) = *(const uint4*)(Ap + (size_t)r * 256 + k0 + c);
    }
#endif
    // stage B [k][n] -> LDS [n][k], k-pairs packed into b32 stores
    int kp = tid >> 3, nc = (tid & 7) * 8;
    const __bf16* s0 = Bp + (size_t)(k0 + 2 * kp) * L_ + nc;
    v8bf d0 = *(const v8bf*)s0;
    v8bf d1 = *(const v8bf*)(s0 + L_);
#pragma unroll
    for (int e = 0; e < 8; ++e) {
      union { __bf16 hh[2]; unsigned u; } p;
      p.hh[0] = d0[e]; p.hh[1] = d1[e];
      *(unsigned*)(Bd + (nc + e) * 48 + 2 * kp) = p.u;
    }
  };

  v8f acc[4] = {};
  stage(0, Asp, Bn);
  for (int kt = 0; kt < 8; ++kt) {
#if __has_builtin(__builtin_amdgcn_tensor_load_to_lds)
    if (wave == 0) __builtin_amdgcn_s_wait_tensorcnt(0);
#endif
    __syncthreads();
    int cura = (kt & 1) * 2048, curb = (kt & 1) * 3072;
    if (kt + 1 < 8)
      stage((kt + 1) << 5, Asp + (((kt + 1) & 1) * 2048), Bn + (((kt + 1) & 1) * 3072));
    // A frag from plain [m][32]: two 16B runs (K=kh..kh+7, kh+8..kh+15), kh=16*(lane>>4)
    const __bf16* rp = Asp + cura + (wave * 16 + (lane & 15)) * 32 + ((lane >> 4) << 4);
    v8bf lo = *(const v8bf*)(rp);
    v8bf hi = *(const v8bf*)(rp + 8);
    v16bf af = __builtin_shufflevector(lo, hi, 0, 1, 2, 3, 4, 5, 6, 7,
                                       8, 9, 10, 11, 12, 13, 14, 15);
#pragma unroll
    for (int nt = 0; nt < 4; ++nt) {
      v16bf bfv = *(const v16bf*)(Bn + curb + (nt * 16 + (lane & 15)) * 48 + ((lane >> 4) << 4));
      acc[nt] = __builtin_amdgcn_wmma_f32_16x16x32_bf16(
          false, af, false, bfv, (short)0, acc[nt], false, false);
    }
    __syncthreads();
  }
#pragma unroll
  for (int nt = 0; nt < 4; ++nt)
#pragma unroll
    for (int r = 0; r < 8; ++r) {
      int c = mb * 64 + frag_row(wave, lane, r);
      int l = h * DV_ + nb * 64 + frag_col(nt, lane);
      ao[(((size_t)b * C_ + c) << 14) + l] = (__bf16)acc[nt][r];
    }
}

// final: gelu(proj@ao + pb)*bn + relu(sc@x + scb) ; two fused WMMA GEMMs
__global__ void k_gemm_final(const __bf16* __restrict__ pw, const __bf16* __restrict__ scw,
                             const __bf16* __restrict__ ao, const float* __restrict__ x,
                             const float* __restrict__ pb, const float* __restrict__ scb,
                             const float* __restrict__ gamma, const float* __restrict__ beta,
                             float* __restrict__ out) {
  __shared__ __attribute__((aligned(32))) __bf16 As[2 * 3072];
  __shared__ __attribute__((aligned(32))) __bf16 Bn[2 * 3072];
  int bid = blockIdx.x;
  int nb = bid & 255, mb = (bid >> 8) & 3, b = bid >> 10;
  v8f acc1[4] = {};
  gemm64x64<__bf16, false>(pw + mb * 64 * 256, 256,
                           ao + (((size_t)b * C_) << 14) + nb * 64, L_, 256,
                           acc1, As, Bn);
  v8f acc2[4] = {};
  gemm64x64<float, false>(scw + mb * 64 * 256, 256,
                          x + (((size_t)b * C_) << 14) + nb * 64, L_, 256,
                          acc2, As, Bn);
  int lane = threadIdx.x & 31, wave = threadIdx.x >> 5;
  const float bnscale = 0.9999950000374997f;  // 1/sqrt(1+1e-5)
#pragma unroll
  for (int nt = 0; nt < 4; ++nt)
#pragma unroll
    for (int r = 0; r < 8; ++r) {
      int o = mb * 64 + frag_row(wave, lane, r);
      int l = nb * 64 + frag_col(nt, lane);
      float y = acc1[nt][r] + pb[o];
      y = 0.5f * y * (1.f + erff(y * 0.70710678118f));     // exact GELU
      y = y * bnscale * gamma[o] + beta[o];                 // BN (inference)
      float sv = fmaxf(acc2[nt][r] + scb[o], 0.f);          // shortcut relu
      out[(((size_t)b * C_ + o) << 14) + l] = y + sv;
    }
}

// ---------------- host launcher ----------------
extern "C" void kernel_launch(void* const* d_in, const int* in_sizes, int n_in,
                              void* d_out, int out_size, void* d_ws, size_t ws_size,
                              hipStream_t stream) {
  (void)in_sizes; (void)n_in; (void)out_size; (void)ws_size;
  const float* x        = (const float*)d_in[0];
  const float* sc_w     = (const float*)d_in[1];
  const float* sc_b     = (const float*)d_in[2];
  const float* q_w      = (const float*)d_in[3];
  const float* q_b      = (const float*)d_in[4];
  const float* k_w      = (const float*)d_in[5];
  const float* k_b      = (const float*)d_in[6];
  const float* v_dw_w   = (const float*)d_in[7];
  const float* v_dw_b   = (const float*)d_in[8];
  const float* v_pw_w   = (const float*)d_in[9];
  const float* v_pw_b   = (const float*)d_in[10];
  const float* logit_sc = (const float*)d_in[11];
  const float* cpb_w1   = (const float*)d_in[12];
  const float* cpb_b1   = (const float*)d_in[13];
  const float* cpb_w2   = (const float*)d_in[14];
  const float* proj_w   = (const float*)d_in[15];
  const float* proj_b   = (const float*)d_in[16];
  const float* bn_gamma = (const float*)d_in[17];
  const float* bn_beta  = (const float*)d_in[18];

  // workspace layout (aliased; ~138.7 MB total)
  char* ws = (char*)d_ws;
  __bf16* v_bf  = (__bf16*)ws;                       // 67,108,864 B (persistent)
  char*   scr   = ws + 67108864;                     // 67 MB scratch, reused 3x
  __bf16* vd_bf = (__bf16*)scr;                      // phase 1: depthwise out
  float*  xp    = (float*)scr;                       // phase 2: q/k pipeline
  float*  qbuf  = (float*)(scr + 8388608);
  float*  kbuf  = (float*)(scr + 16777216);
  float*  simb  = (float*)(scr + 25165824);
  float*  biasb = (float*)(scr + 33554432);
  __bf16* qn    = (__bf16*)(scr + 34603008);
  __bf16* kn    = (__bf16*)(scr + 38797312);
  __bf16* ao    = (__bf16*)scr;                      // phase 3: attention out
  __bf16* pw_bf = (__bf16*)(ws + 134217728);         // weights (294,912 B)
  __bf16* sc_bf = pw_bf + 65536;
  __bf16* vpwbf = pw_bf + 131072;
  __bf16* attn  = (__bf16*)(ws + 134512640);         // 4,194,304 B

  // weights -> bf16
  k_f32bf16<<<256, 256, 0, stream>>>(proj_w, pw_bf, 65536);
  k_f32bf16<<<256, 256, 0, stream>>>(sc_w, sc_bf, 65536);
  k_f32bf16<<<64, 256, 0, stream>>>(v_pw_w, vpwbf, 16384);

  // v path first (vd occupies the scratch region, then dies)
  k_dwconv<<<(BN_ * C_ * L_) / 256, 256, 0, stream>>>(x, v_dw_w, v_dw_b, vd_bf);
  k_gemm_v<<<32 * 256, 128, 0, stream>>>(vpwbf, vd_bf, v_pw_b, v_bf);

  // q/k path (reuses scratch)
  k_bias<<<256, 256, 0, stream>>>(cpb_w1, cpb_b1, cpb_w2, biasb);
  k_maxpool<<<(BN_ * C_ * LP_) / 256, 256, 0, stream>>>(x, xp);
  k_qk<<<(BN_ * C_ * LP_) / 256, 256, 0, stream>>>(xp, q_w, q_b, k_w, k_b, qbuf, kbuf);
  k_rownorm<<<1024, 256, 0, stream>>>(qbuf, kbuf, qn, kn);
  k_gemm_sim<<<512, 128, 0, stream>>>(qn, kn, biasb, logit_sc, simb);
  k_softmax2<<<1024, 256, 0, stream>>>(simb, attn);

  // attention applied to v (ao reuses scratch; q/k buffers dead)
  k_gemm_attnout<<<8192, 128, 0, stream>>>(attn, v_bf, ao);

  // fused proj(GELU,BN) + shortcut(ReLU) + add
  k_gemm_final<<<8192, 128, 0, stream>>>(pw_bf, sc_bf, ao, x, proj_b, sc_b,
                                         bn_gamma, bn_beta, (float*)d_out);
}
